// Model_41059887350268
// MI455X (gfx1250) — compile-verified, matrix operand order checked by reading, not run on hardware
//
#include <hip/hip_runtime.h>
#include <math.h>
#include <stdint.h>

#define N_NODES   200000
#define IN_DIM_C  128
#define HID_C     64
#define E_COMPLETE 600000
#define E_SPLIT    500000
#define E_LABEL    400000
#define HASH_BITS  21
#define HASH_CAP   (1u << HASH_BITS)
#define HASH_MASK  (HASH_CAP - 1u)

typedef __attribute__((ext_vector_type(16))) __bf16 v16bf;
typedef __attribute__((ext_vector_type(8)))  __bf16 v8bf;
typedef __attribute__((ext_vector_type(8)))  float  v8f;

// ---------------- device helpers ----------------
__device__ __forceinline__ float dev_sigmoid(float x) { return 1.0f / (1.0f + __expf(-x)); }
__device__ __forceinline__ float dev_gelu(float x)    { return 0.5f * x * (1.0f + erff(x * 0.70710678118654752f)); }

__device__ __forceinline__ unsigned mix64(unsigned long long k) {
  k ^= k >> 33; k *= 0xff51afd7ed558ccdULL;
  k ^= k >> 33; k *= 0xc4ceb9fe1a85ec53ULL;
  k ^= k >> 33;
  return (unsigned)k;
}

__device__ __forceinline__ void atomicMaxF(float* addr, float val) {
  if (val >= 0.0f) atomicMax((int*)addr, __float_as_int(val));
  else             atomicMin((unsigned int*)addr, __float_as_uint(val));
}

// ---------------- utility kernels ----------------
__global__ void k_fill(float* __restrict__ p, float v, long n) {
  long i = (long)blockIdx.x * blockDim.x + threadIdx.x;
  if (i < n) p[i] = v;
}

__global__ void k_hash_init(unsigned long long* __restrict__ keys) {
  unsigned i = blockIdx.x * blockDim.x + threadIdx.x;
  if (i < HASH_CAP) keys[i] = ~0ULL;
}

__global__ void k_hash_build(const int* __restrict__ edges,
                             unsigned long long* __restrict__ keys,
                             int* __restrict__ vals) {
  int e = blockIdx.x * blockDim.x + threadIdx.x;
  if (e >= E_COMPLETE) return;
  unsigned long long key =
      (unsigned long long)(unsigned)edges[e] * (unsigned long long)N_NODES +
      (unsigned long long)(unsigned)edges[E_COMPLETE + e];
  unsigned h = mix64(key) & HASH_MASK;
  for (unsigned probe = 0; probe < HASH_CAP; ++probe) {
    unsigned long long prev = atomicCAS(&keys[h], ~0ULL, key);
    if (prev == ~0ULL || prev == key) { vals[h] = e; break; }
    h = (h + 1u) & HASH_MASK;
  }
}

__global__ void k_hash_lookup(const int* __restrict__ sedges,
                              const unsigned long long* __restrict__ keys,
                              const int* __restrict__ vals,
                              const float* __restrict__ ew,
                              float* __restrict__ relw) {
  int e = blockIdx.x * blockDim.x + threadIdx.x;
  if (e >= E_SPLIT) return;
  unsigned long long key =
      (unsigned long long)(unsigned)sedges[e] * (unsigned long long)N_NODES +
      (unsigned long long)(unsigned)sedges[E_SPLIT + e];
  unsigned h = mix64(key) & HASH_MASK;
  int idx = 0;
  for (unsigned probe = 0; probe < HASH_CAP; ++probe) {
    unsigned long long k = keys[h];
    if (k == key) { idx = vals[h]; break; }
    if (k == ~0ULL) break;   // not found (shouldn't happen)
    h = (h + 1u) & HASH_MASK;
  }
  relw[e] = dev_sigmoid(ew[idx]);
}

// one winner per row (matches XLA scatter-set nondeterminism for duplicate rows)
__global__ void k_scale_set(const int* __restrict__ sedges,
                            const float* __restrict__ relw,
                            float* __restrict__ scalebuf) {
  int e = blockIdx.x * blockDim.x + threadIdx.x;
  if (e >= E_SPLIT) return;
  scalebuf[sedges[e]] = relw[e];
}

__global__ void k_scale_apply(float* __restrict__ xd,
                              const float* __restrict__ scalebuf) {
  long i = (long)blockIdx.x * blockDim.x + threadIdx.x;
  if (i >= (long)N_NODES * HID_C) return;
  xd[i] *= scalebuf[i >> 6];
}

// build 64x64 block-diagonal weight from (2,32,32) per-head matrices
__global__ void k_blockdiag(const float* __restrict__ a, float* __restrict__ wd) {
  int t = blockIdx.x * blockDim.x + threadIdx.x;
  if (t >= 64 * 64) return;
  int r = t >> 6, c = t & 63;
  int hr = r >> 5, hc = c >> 5;
  wd[t] = (hr == hc) ? a[(hr << 10) + ((r & 31) << 5) + (c & 31)] : 0.0f;
}

// ---------------- WMMA GEMM ----------------
// C[M x NT*16] = act_out( act_in(A)[M x K] @ W[K x NT*16] + bias ), optional skip.
// One wave per 16-row M-tile computing ALL NT column tiles (A loaded/converted once).
// W panel (contiguous K x NOUT f32) is DMA'd into LDS with
// GLOBAL_LOAD_ASYNC_TO_LDS_B128 (ASYNCcnt), drained with s_wait_asynccnt, then
// converted once into a transposed bf16 panel so each B fragment is two
// contiguous ds_load_b128 reads. flags: 1=GELU on A, 2=ReLU out, 4=skip blend.
template <int K, int NT>
__global__ __launch_bounds__(256)
void k_wmma_gemm(const float* __restrict__ A, int lda,
                 const float* __restrict__ W,
                 const float* __restrict__ bias,
                 float* __restrict__ C, int ldc,
                 int M, int flags,
                 const float* __restrict__ skipx,
                 const float* __restrict__ skipg) {
  constexpr int NOUT = NT * 16;
  __shared__ float  wf[K * NOUT];      // async-staged f32 panel [K][NOUT]
  __shared__ __bf16 wT[NOUT * K];      // transposed bf16 panel [NOUT][K]

  // --- async DMA stage of the weight panel (16B per lane per issue) ---
  for (int i = threadIdx.x * 4; i < K * NOUT; i += blockDim.x * 4) {
    unsigned lds_off = (unsigned)(uintptr_t)(&wf[i]);            // addr[31:0] = LDS offset
    unsigned long long gaddr = (unsigned long long)(uintptr_t)(W + i);
    asm volatile("global_load_async_to_lds_b128 %0, %1, off"
                 :: "v"(lds_off), "v"(gaddr) : "memory");
  }
  asm volatile("s_wait_asynccnt 0" ::: "memory");
  __syncthreads();
  // --- one-time f32 -> transposed bf16 conversion in LDS ---
  for (int idx = threadIdx.x; idx < K * NOUT; idx += blockDim.x) {
    int n = idx / K, k = idx % K;
    wT[idx] = (__bf16)wf[k * NOUT + n];
  }
  __syncthreads();

  int lane = threadIdx.x & 31;
  long wave = (long)blockIdx.x * (blockDim.x >> 5) + (threadIdx.x >> 5);
  int mtiles = (M + 15) >> 4;
  if (wave >= mtiles) return;
  int tM   = (int)wave;
  int half = lane >> 4;      // 0: K 0-7/16-23, 1: K 8-15/24-31
  int l    = lane & 15;

  int row = tM * 16 + l;
  if (row >= M) row = M - 1;   // M=200000 is 16-divisible; defensive only
  const float* arow = A + (long)row * lda + half * 8;

  v8f acc[NT];
#pragma unroll
  for (int t = 0; t < NT; ++t) acc[t] = (v8f){0.f, 0.f, 0.f, 0.f, 0.f, 0.f, 0.f, 0.f};

#pragma unroll
  for (int k0 = 0; k0 < K; k0 += 32) {
    v16bf af;
#pragma unroll
    for (int i = 0; i < 8; ++i) {
      float x0 = arow[k0 + i];
      float x1 = arow[k0 + 16 + i];
      if (flags & 1) { x0 = dev_gelu(x0); x1 = dev_gelu(x1); }
      af[i]     = (__bf16)x0;
      af[8 + i] = (__bf16)x1;
    }
#pragma unroll
    for (int t = 0; t < NT; ++t) {
      const __bf16* wp = &wT[(t * 16 + l) * K + k0 + half * 8];
      v8bf lo = *(const v8bf*)wp;         // ds_load_b128
      v8bf hi = *(const v8bf*)(wp + 16);  // ds_load_b128
      v16bf bf = __builtin_shufflevector(lo, hi, 0, 1, 2, 3, 4, 5, 6, 7,
                                                 8, 9, 10, 11, 12, 13, 14, 15);
      acc[t] = __builtin_amdgcn_wmma_f32_16x16x32_bf16(
          false, af, false, bf, (short)0, acc[t], false, false);
    }
  }

  float sg = 0.0f, sgc = 0.0f;
  if (flags & 4) { float tv = *skipg; sg = dev_sigmoid(tv); sgc = 1.0f - sg; }
#pragma unroll
  for (int t = 0; t < NT; ++t) {
    int n = t * 16 + l;
    float bv = bias ? bias[n] : 0.0f;
#pragma unroll
    for (int i = 0; i < 8; ++i) {
      int m = tM * 16 + half * 8 + i;
      if (m < M) {
        float o = acc[t][i] + bv;
        if (flags & 2) o = fmaxf(o, 0.0f);
        if (flags & 4) o = sg * o + sgc * skipx[(long)m * ldc + n];
        C[(long)m * ldc + n] = o;
      }
    }
  }
}

// ---------------- edge attention kernels (per relation) ----------------
__global__ void k_edge_logits(const int* __restrict__ se,
                              const float* __restrict__ kqv_dst,  // q at col 64, stride 192
                              const float* __restrict__ ktr,      // N x 64
                              const float* __restrict__ prel,     // 2
                              float* __restrict__ logits,         // E x 2
                              float* __restrict__ mbuf) {         // N x 2
  int tid = blockIdx.x * blockDim.x + threadIdx.x;
  if (tid >= E_SPLIT * 2) return;
  int e = tid >> 1, h = tid & 1;
  int row = se[e], col = se[E_SPLIT + e];
  const float4* q = (const float4*)(kqv_dst + (long)col * 192 + 64 + h * 32);
  const float4* k = (const float4*)(ktr + (long)row * 64 + h * 32);
  float acc = 0.0f;
#pragma unroll
  for (int i = 0; i < 8; ++i) {
    float4 qv = q[i], kv = k[i];
    acc += qv.x * kv.x + qv.y * kv.y + qv.z * kv.z + qv.w * kv.w;
  }
  float lg = acc * prel[h] * 0.17677669529663689f;  // 1/sqrt(32)
  logits[tid] = lg;
  atomicMaxF(&mbuf[(long)col * 2 + h], lg);
}

__global__ void k_edge_expsum(const int* __restrict__ se,
                              float* __restrict__ logits,
                              const float* __restrict__ mbuf,
                              float* __restrict__ sbuf) {
  int tid = blockIdx.x * blockDim.x + threadIdx.x;
  if (tid >= E_SPLIT * 2) return;
  int e = tid >> 1, h = tid & 1;
  int col = se[E_SPLIT + e];
  float ex = __expf(logits[tid] - mbuf[(long)col * 2 + h]);
  logits[tid] = ex;
  atomicAdd(&sbuf[(long)col * 2 + h], ex);
}

__global__ void k_edge_scatter(const int* __restrict__ se,
                               const float* __restrict__ logits,
                               const float* __restrict__ sbuf,
                               const float* __restrict__ vtr,  // N x 64
                               float* __restrict__ agg) {      // N x 64
  int tid = blockIdx.x * blockDim.x + threadIdx.x;
  if (tid >= E_SPLIT * 2) return;
  int e = tid >> 1, h = tid & 1;
  int row = se[e], col = se[E_SPLIT + e];
  float alpha = logits[tid] / sbuf[(long)col * 2 + h];
  const float* vv = vtr + (long)row * 64 + h * 32;
  float* ag = agg + (long)col * 64 + h * 32;
#pragma unroll
  for (int d = 0; d < 32; ++d) atomicAdd(&ag[d], alpha * vv[d]);
}

// ---------------- prediction ----------------
__global__ void k_predict(const int* __restrict__ eli,
                          const float* __restrict__ zq,
                          const float* __restrict__ za,
                          float* __restrict__ pred) {
  int e = blockIdx.x * blockDim.x + threadIdx.x;
  if (e >= E_LABEL) return;
  int i0 = eli[e], i1 = eli[E_LABEL + e];
  const float4* a = (const float4*)(zq + (long)i0 * 64);
  const float4* b = (const float4*)(za + (long)i1 * 64);
  float acc = 0.0f;
#pragma unroll
  for (int i = 0; i < 16; ++i) {
    float4 av = a[i], bv = b[i];
    acc += av.x * bv.x + av.y * bv.y + av.z * bv.z + av.w * bv.w;
  }
  pred[e] = acc;
}

// ---------------- host orchestration ----------------
static inline unsigned gdiv(long n, int b) { return (unsigned)((n + b - 1) / b); }

extern "C" void kernel_launch(void* const* d_in, const int* in_sizes, int n_in,
                              void* d_out, int out_size, void* d_ws, size_t ws_size,
                              hipStream_t stream) {
  (void)in_sizes; (void)n_in; (void)out_size;
  // ---- inputs (setup_inputs dict order, depth-first) ----
  const float* x_in[3]  = {(const float*)d_in[0], (const float*)d_in[1], (const float*)d_in[2]};
  const int*   ce[3]    = {(const int*)d_in[3], (const int*)d_in[4], (const int*)d_in[5]};
  const int*   se[3]    = {(const int*)d_in[6], (const int*)d_in[7], (const int*)d_in[8]};
  const int*   eli      = (const int*)d_in[9];
  const float* lin_w[3] = {(const float*)d_in[10], (const float*)d_in[12], (const float*)d_in[14]};
  const float* lin_b[3] = {(const float*)d_in[11], (const float*)d_in[13], (const float*)d_in[15]};
  const float* ew[3]    = {(const float*)d_in[16], (const float*)d_in[17], (const float*)d_in[18]};
  const float* kqv_w[3] = {(const float*)d_in[19], (const float*)d_in[21], (const float*)d_in[23]};
  const float* kqv_b[3] = {(const float*)d_in[20], (const float*)d_in[22], (const float*)d_in[24]};
  const float* out_w[3] = {(const float*)d_in[25], (const float*)d_in[27], (const float*)d_in[29]};
  const float* out_b[3] = {(const float*)d_in[26], (const float*)d_in[28], (const float*)d_in[30]};
  const float* skipg[3] = {(const float*)d_in[31], (const float*)d_in[32], (const float*)d_in[33]};
  const float* a_rel[3] = {(const float*)d_in[34], (const float*)d_in[35], (const float*)d_in[36]};
  const float* m_rel[3] = {(const float*)d_in[37], (const float*)d_in[38], (const float*)d_in[39]};
  const float* p_rel[3] = {(const float*)d_in[40], (const float*)d_in[41], (const float*)d_in[42]};
  float* pred = (float*)d_out;

  // ---- workspace carve-out ----
  char* ws = (char*)d_ws;
  size_t off = 0;
  auto carve = [&](size_t bytes) -> void* {
    void* p = ws + off;
    off += (bytes + 255) & ~(size_t)255;
    return p;
  };
  const long NF = (long)N_NODES * HID_C;       // 12.8M floats
  float* xd[3];  for (int t = 0; t < 3; ++t) xd[t]  = (float*)carve(NF * 4);
  float* kqv[3]; for (int t = 0; t < 3; ++t) kqv[t] = (float*)carve((long)N_NODES * 192 * 4);
  float* ktr   = (float*)carve(NF * 4);
  float* vtr   = (float*)carve(NF * 4);
  float* agg_q = (float*)carve(NF * 4);
  float* agg_a = (float*)carve(NF * 4);
  float* z_q   = (float*)carve(NF * 4);
  float* z_a   = (float*)carve(NF * 4);
  float* relw[3]; for (int r = 0; r < 3; ++r) relw[r] = (float*)carve((long)E_SPLIT * 4);
  unsigned long long* hkeys = (unsigned long long*)carve((size_t)HASH_CAP * 8);
  int*   hvals    = (int*)carve((size_t)HASH_CAP * 4);
  float* scalebuf = (float*)carve((long)N_NODES * 4);
  float* logits   = (float*)carve((long)E_SPLIT * 2 * 4);
  float* mbuf     = (float*)carve((long)N_NODES * 2 * 4);
  float* sbuf     = (float*)carve((long)N_NODES * 2 * 4);
  float* wdiag    = (float*)carve(64 * 64 * 4);
  if (off > ws_size) return;  // insufficient scratch; do nothing (deterministic)

  const int TB = 256;
  const unsigned GB = gdiv(((N_NODES + 15) / 16), 8);  // 1 M-tile per wave, 8 waves/block

  // Phase 1: match_indices via hash join -> relw[r] = sigmoid(edge_weights[match])
  for (int r = 0; r < 3; ++r) {
    k_hash_init<<<gdiv(HASH_CAP, TB), TB, 0, stream>>>(hkeys);
    k_hash_build<<<gdiv(E_COMPLETE, TB), TB, 0, stream>>>(ce[r], hkeys, hvals);
    k_hash_lookup<<<gdiv(E_SPLIT, TB), TB, 0, stream>>>(se[r], hkeys, hvals, ew[r], relw[r]);
  }

  // Phase 2: xd[t] = relu(x[t] @ lin_w + lin_b)   (WMMA, K=128 -> 64)
  for (int t = 0; t < 3; ++t)
    k_wmma_gemm<IN_DIM_C, 4><<<GB, TB, 0, stream>>>(
        x_in[t], IN_DIM_C, lin_w[t], lin_b[t], xd[t], HID_C, N_NODES, /*relu*/2, nullptr, nullptr);

  // Phase 3: edge-weight row scaling of wx (question by qa then qc; concept by ca)
  const int   scale_rel[3] = {0, 1, 2};
  float*      scale_tgt[3] = {xd[0], xd[0], xd[2]};
  for (int s = 0; s < 3; ++s) {
    k_fill<<<gdiv(N_NODES, TB), TB, 0, stream>>>(scalebuf, 1.0f, N_NODES);
    k_scale_set<<<gdiv(E_SPLIT, TB), TB, 0, stream>>>(se[scale_rel[s]], relw[scale_rel[s]], scalebuf);
    k_scale_apply<<<gdiv(NF, TB), TB, 0, stream>>>(scale_tgt[s], scalebuf);
  }

  // Phase 4: kqv[t] = wx[t] @ kqv_w + kqv_b   (WMMA, K=64 -> 192, 12 N-tiles/wave)
  for (int t = 0; t < 3; ++t)
    k_wmma_gemm<HID_C, 12><<<GB, TB, 0, stream>>>(
        xd[t], HID_C, kqv_w[t], kqv_b[t], kqv[t], 192, N_NODES, 0, nullptr, nullptr);

  // Phase 5: zero aggregation buffers (question has no incoming edges -> stays 0)
  k_fill<<<gdiv(NF, TB), TB, 0, stream>>>(agg_q, 0.0f, NF);
  k_fill<<<gdiv(NF, TB), TB, 0, stream>>>(agg_a, 0.0f, NF);

  // Phase 6: edge attention. Only relations with dst='answer' reach the output
  // (z_concept is dead; question is never a dst), so run qa (r=0) and ca (r=2).
  const int run_rel[2] = {0, 2};
  const int src_ty[2]  = {0, 2};  // question, concept
  for (int i = 0; i < 2; ++i) {
    int r = run_rel[i];
    const float* skqv = kqv[src_ty[i]];
    // k_rel = k[src] @ blockdiag(a_rel)  (WMMA)
    k_blockdiag<<<gdiv(64 * 64, TB), TB, 0, stream>>>(a_rel[r], wdiag);
    k_wmma_gemm<HID_C, 4><<<GB, TB, 0, stream>>>(
        skqv + 0, 192, wdiag, nullptr, ktr, HID_C, N_NODES, 0, nullptr, nullptr);
    // v_rel = v[src] @ blockdiag(m_rel)  (WMMA)
    k_blockdiag<<<gdiv(64 * 64, TB), TB, 0, stream>>>(m_rel[r], wdiag);
    k_wmma_gemm<HID_C, 4><<<GB, TB, 0, stream>>>(
        skqv + 128, 192, wdiag, nullptr, vtr, HID_C, N_NODES, 0, nullptr, nullptr);
    // segment softmax over dst='answer' + weighted scatter into agg_a
    k_fill<<<gdiv((long)N_NODES * 2, TB), TB, 0, stream>>>(mbuf, -INFINITY, (long)N_NODES * 2);
    k_fill<<<gdiv((long)N_NODES * 2, TB), TB, 0, stream>>>(sbuf, 0.0f, (long)N_NODES * 2);
    k_edge_logits<<<gdiv((long)E_SPLIT * 2, TB), TB, 0, stream>>>(se[r], kqv[1], ktr, p_rel[r], logits, mbuf);
    k_edge_expsum<<<gdiv((long)E_SPLIT * 2, TB), TB, 0, stream>>>(se[r], logits, mbuf, sbuf);
    k_edge_scatter<<<gdiv((long)E_SPLIT * 2, TB), TB, 0, stream>>>(se[r], logits, sbuf, vtr, agg_a);
  }

  // Phase 7: z[t] = sig(skip)*(gelu(agg) @ out_w + out_b) + (1-sig(skip))*wx[t]
  k_wmma_gemm<HID_C, 4><<<GB, TB, 0, stream>>>(
      agg_q, HID_C, out_w[0], out_b[0], z_q, HID_C, N_NODES, 1 | 4, xd[0], skipg[0]);
  k_wmma_gemm<HID_C, 4><<<GB, TB, 0, stream>>>(
      agg_a, HID_C, out_w[1], out_b[1], z_a, HID_C, N_NODES, 1 | 4, xd[1], skipg[1]);

  // Phase 8: pred[e] = dot(z_q[eli0], z_a[eli1])
  k_predict<<<gdiv(E_LABEL, TB), TB, 0, stream>>>(eli, z_q, z_a, pred);
}
